// GraphAttention_62981400429165
// MI455X (gfx1250) — matrix operand order repaired and checked by reading, not compiled, not run on hardware
//
#include <hip/hip_runtime.h>

#define BB 4
#define NN 2048
#define FIN 256
#define HH 8
#define FO 64
#define NEG_BIG -1.0e10f
#define ALPHA 0.2f

typedef __attribute__((ext_vector_type(16))) __bf16 v16bf;
typedef __attribute__((ext_vector_type(8)))  float v8f;

union Frag16 {
    uint4 q[2];
    unsigned short h[16];
    v16bf v;
};

__device__ inline unsigned short f32_bf16(float f) {
    unsigned u = __float_as_uint(f);
    unsigned r = 0x7FFFu + ((u >> 16) & 1u);
    return (unsigned short)((u + r) >> 16);
}

// ---------------- Stage 0: precision conversion ----------------
__global__ void cvt_x_kernel(const float* __restrict__ X,
                             unsigned short* __restrict__ Xb, int n) {
    int i = blockIdx.x * blockDim.x + threadIdx.x;
    if (i < n) Xb[i] = f32_bf16(X[i]);
}

// W [H,F,FO] -> WT bf16 [H,FO,F]
__global__ void cvt_w_kernel(const float* __restrict__ W,
                             unsigned short* __restrict__ WT) {
    int i = blockIdx.x * blockDim.x + threadIdx.x;
    if (i >= HH * FIN * FO) return;
    int o = i % FO;
    int f = (i / FO) % FIN;
    int h = i / (FO * FIN);
    WT[((size_t)h * FO + o) * FIN + f] = f32_bf16(W[i]);
}

// ---------------- Stage 1: h = X @ W (per head), att_s / att_n ----------------
// One wave per (b, head, 16-row tile). Stores h transposed bf16: hT[bh][o][n].
__global__ __launch_bounds__(128)
void gat_stage1(const unsigned short* __restrict__ Xb,
                const unsigned short* __restrict__ WT,
                const float* __restrict__ a_self,
                const float* __restrict__ a_neigh,
                unsigned short* __restrict__ hT,
                float* __restrict__ att_s,
                float* __restrict__ att_n) {
    const int lane = threadIdx.x & 31;
    const int wave = threadIdx.x >> 5;
    const int wid  = blockIdx.x * 4 + wave;     // 0..4095
    const int bh   = wid >> 7;                  // 0..31 (b*8+head)
    const int tile = (wid & 127) << 4;          // row tile base
    const int b    = bh >> 3;
    const int head = bh & 7;

    const int lrow  = lane & 15;
    const int half  = lane >> 4;
    const int off8  = half * 8;
    const int off16 = half * 16;

    v8f acc[4] = {};

    const unsigned short* xrow = Xb + ((size_t)b * NN + (tile + lrow)) * FIN;
    const unsigned short* wt   = WT + (size_t)head * FO * FIN;

    for (int kk = 0; kk < FIN; kk += 32) {
        Frag16 afr;
        afr.q[0] = *(const uint4*)(xrow + kk + off8);       // j=0..7  : f=kk+off8+j
        afr.q[1] = *(const uint4*)(xrow + kk + 16 + off8);  // j=8..15 : f=kk+16+off8+(j-8)
#pragma unroll
        for (int t = 0; t < 4; ++t) {
            Frag16 bfr;
            const unsigned short* wcol =
                wt + (size_t)(t * 16 + lrow) * FIN + kk + off8;
            bfr.q[0] = *(const uint4*)(wcol);       // j=0..7  : f=kk+off8+j
            bfr.q[1] = *(const uint4*)(wcol + 16);  // j=8..15 : f=kk+16+off8+(j-8)
            acc[t] = __builtin_amdgcn_wmma_f32_16x16x32_bf16(
                false, afr.v, false, bfr.v, (short)0, acc[t], false, false);
        }
    }

    // attention logits: dot(h_row, a_self/a_neigh), reduced within each 16-lane half
    float as[4], an[4];
#pragma unroll
    for (int t = 0; t < 4; ++t) {
        as[t] = a_self [head * FO + t * 16 + lrow];
        an[t] = a_neigh[head * FO + t * 16 + lrow];
    }
#pragma unroll
    for (int r = 0; r < 8; ++r) {
        float ps = 0.f, pn = 0.f;
#pragma unroll
        for (int t = 0; t < 4; ++t) {
            float v = acc[t][r];
            ps += v * as[t];
            pn += v * an[t];
        }
#pragma unroll
        for (int m = 1; m < 16; m <<= 1) {
            ps += __shfl_xor(ps, m, 32);
            pn += __shfl_xor(pn, m, 32);
        }
        if (lrow == 0) {
            int row = tile + r + half * 8;
            att_s[(size_t)bh * NN + row] = ps;
            att_n[(size_t)bh * NN + row] = pn;
        }
    }

    // store hT bf16: lane owns column o, rows tile+half*8 .. +7 contiguous -> one b128 per o-tile
#pragma unroll
    for (int t = 0; t < 4; ++t) {
        unsigned pk[4];
#pragma unroll
        for (int p = 0; p < 4; ++p) {
            pk[p] = (unsigned)f32_bf16(acc[t][2 * p]) |
                    ((unsigned)f32_bf16(acc[t][2 * p + 1]) << 16);
        }
        uint4 q = make_uint4(pk[0], pk[1], pk[2], pk[3]);
        int o = t * 16 + lrow;
        *(uint4*)(hT + ((size_t)bh * FO + o) * NN + tile + half * 8) = q;
    }
}

// ---------------- Stage 2: fused masked softmax + aggregation (flash style) ----------------
// One wave per (b, head, 16-row tile); online softmax over 64 blocks of 32 columns.
__global__ __launch_bounds__(128)
void gat_stage2(const float* __restrict__ A,
                const unsigned short* __restrict__ hT,
                const float* __restrict__ att_s,
                const float* __restrict__ att_n,
                float* __restrict__ out) {
    const int lane = threadIdx.x & 31;
    const int wave = threadIdx.x >> 5;
    const int wid  = blockIdx.x * 4 + wave;
    const int bh   = wid >> 7;
    const int tile = (wid & 127) << 4;
    const int b    = bh >> 3;
    const int head = bh & 7;

    const int lrow  = lane & 15;
    const int half  = lane >> 4;
    const int off8  = half * 8;
    const int off16 = half * 16;

    const int n = tile + lrow;                       // row this lane scores (A-frag layout)
    const float asv = att_s[(size_t)bh * NN + n];
    const float* Arow  = A + ((size_t)b * NN + n) * NN;
    const float* anrow = att_n + (size_t)bh * NN;
    const unsigned short* hbase = hT + (size_t)bh * FO * NN;

    v8f O[4] = {};
    float M = -INFINITY, L = 0.0f;

    for (int mb = 0; mb < NN; mb += 32) {
        if (mb + 32 < NN) __builtin_prefetch(Arow + mb + 32 + off8, 0, 0);

        float s[16];
        float lm = -INFINITY;
#pragma unroll
        for (int c = 0; c < 4; ++c) {
            const int coff = ((c & 1) ? 4 : 0) + ((c >> 1) ? 16 : 0) + off8;
            float4 qa = *(const float4*)(Arow  + mb + coff);
            float4 qn = *(const float4*)(anrow + mb + coff);
            float aa[4] = {qa.x, qa.y, qa.z, qa.w};
            float nn4[4] = {qn.x, qn.y, qn.z, qn.w};
#pragma unroll
            for (int e = 0; e < 4; ++e) {
                float raw = asv + nn4[e];
                float sc = raw > 0.f ? raw : ALPHA * raw;       // LeakyReLU(0.2)
                sc = (aa[e] == 0.f) ? sc + NEG_BIG : sc;        // adjacency mask
                s[c * 4 + e] = sc;
                lm = fmaxf(lm, sc);
            }
        }
        // combine row halves (lane ^ 16 holds the other 16 columns of the same row)
        lm = fmaxf(lm, __shfl_xor(lm, 16, 32));
        float Mnew = fmaxf(M, lm);
        float c0 = __expf(M - Mnew);

        float ls = 0.f;
        Frag16 P;
#pragma unroll
        for (int j = 0; j < 16; ++j) {
            float p = __expf(s[j] - Mnew);
            ls += p;
            P.h[j] = f32_bf16(p);
        }
        ls += __shfl_xor(ls, 16, 32);
        L = L * c0 + ls;
        M = Mnew;

        // rescale accumulator: O-frag row = r + 8*half; its scale lives at lane (r + 8*half)
#pragma unroll
        for (int r = 0; r < 8; ++r) {
            float cr = __shfl(c0, r + half * 8, 32);
#pragma unroll
            for (int t = 0; t < 4; ++t) O[t][r] *= cr;
        }

        // O += P @ h  (B-frag: lane owns o-column, K=mb+off8+j for j<8, mb+16+off8+(j-8) for j>=8)
#pragma unroll
        for (int t = 0; t < 4; ++t) {
            Frag16 Bf;
            const unsigned short* hcol =
                hbase + (size_t)(t * 16 + lrow) * NN + mb + off8;
            Bf.q[0] = *(const uint4*)(hcol);
            Bf.q[1] = *(const uint4*)(hcol + 16);
            O[t] = __builtin_amdgcn_wmma_f32_16x16x32_bf16(
                false, P.v, false, Bf.v, (short)0, O[t], false, false);
        }
    }

    // normalize, ReLU, concat-store: out[b, n, head*64 + o]
    float linv = 1.0f / L;
#pragma unroll
    for (int r = 0; r < 8; ++r) {
        float lr = __shfl(linv, r + half * 8, 32);
        int row = tile + r + half * 8;
        float* orow = out + ((size_t)(b * NN + row)) * (HH * FO) + head * FO;
#pragma unroll
        for (int t = 0; t < 4; ++t) {
            float v = O[t][r] * lr;
            orow[t * 16 + lrow] = v > 0.f ? v : 0.f;
        }
    }
}

// ---------------- launch ----------------
extern "C" void kernel_launch(void* const* d_in, const int* in_sizes, int n_in,
                              void* d_out, int out_size, void* d_ws, size_t ws_size,
                              hipStream_t stream) {
    const float* X       = (const float*)d_in[0];
    const float* A       = (const float*)d_in[1];
    const float* W       = (const float*)d_in[2];
    const float* a_self  = (const float*)d_in[3];
    const float* a_neigh = (const float*)d_in[4];
    float* out = (float*)d_out;

    char* ws = (char*)d_ws;
    unsigned short* hT  = (unsigned short*)(ws);                         // 8 MB  [B*H, FO, N] bf16
    unsigned short* Xb  = (unsigned short*)(ws + (size_t)8  * 1024 * 1024); // 4 MB  [B,N,F] bf16
    unsigned short* WT  = (unsigned short*)(ws + (size_t)12 * 1024 * 1024); // 256KB [H,FO,F] bf16
    float* att_s = (float*)(ws + (size_t)12 * 1024 * 1024 + 256 * 1024);    // 256KB [B*H,N]
    float* att_n = (float*)(ws + (size_t)12 * 1024 * 1024 + 512 * 1024);    // 256KB [B*H,N]

    const int nx = BB * NN * FIN;           // 2,097,152
    cvt_x_kernel<<<(nx + 255) / 256, 256, 0, stream>>>(X, Xb, nx);
    const int nw = HH * FIN * FO;           // 131,072
    cvt_w_kernel<<<(nw + 255) / 256, 256, 0, stream>>>(W, WT);

    // 4 waves/block (wave32), one wave per (b, head, 16-row tile): 4*8*128 = 4096 waves
    gat_stage1<<<1024, 128, 0, stream>>>(Xb, WT, a_self, a_neigh, hT, att_s, att_n);
    gat_stage2<<<1024, 128, 0, stream>>>(A, hT, att_s, att_n, out);
}
